// Decoder_78597901517412
// MI455X (gfx1250) — matrix operand order repaired and checked by reading, not compiled, not run on hardware
//
#include <hip/hip_runtime.h>
#include <hip/hip_bf16.h>

#define VOCAB   32000
#define EMB     100
#define UNITS   1024
#define BATCH   128
#define TLEN    512
#define NGATE   4
#define ZDIM    (NGATE * UNITS)   // 4096
#define NWG     64                // one per 16-unit slice
#define NTHREAD 256               // 8 wave32s
#define USE_ASYNC 1

typedef __attribute__((ext_vector_type(16))) __bf16       v16bf;
typedef __attribute__((ext_vector_type(8)))  float        v8f;
typedef __attribute__((ext_vector_type(8)))  unsigned int v8u;

static __device__ __forceinline__ unsigned short f2bf(float f) {
    unsigned u = __float_as_uint(f);
    u += 0x7FFFu + ((u >> 16) & 1u);       // round-to-nearest-even
    return (unsigned short)(u >> 16);
}
static __device__ __forceinline__ float sigf(float x) {
    return 1.0f / (1.0f + __expf(-x));
}
// Byte offset of a shared-memory pointer relative to the wave's LDS base.
static __device__ __forceinline__ unsigned lds_off(const void* p) {
    return (unsigned)(size_t)(const __attribute__((address_space(3))) char*)p;
}

// Load a 16x32 bf16 A-fragment from a row-major [rows x ldk] bf16 matrix.
// Lanes 0-15: row M=lane, K = {kb..kb+7, kb+16..kb+23}, kb=kt*32;
// lanes 16-31: same rows, kb = kt*32+8.
static __device__ __forceinline__ v16bf load_a16(const unsigned short* base,
                                                 int row, int ldk, int kt, int lane) {
    int kb = kt * 32 + ((lane & 16) ? 8 : 0);
    const unsigned short* p = base + (size_t)row * ldk + kb;
    uint4 lo = *(const uint4*)(p);        // K = kb .. kb+7
    uint4 hi = *(const uint4*)(p + 16);   // K = kb+16 .. kb+23
    v8u u;
    u[0] = lo.x; u[1] = lo.y; u[2] = lo.z; u[3] = lo.w;
    u[4] = hi.x; u[5] = hi.y; u[6] = hi.z; u[7] = hi.w;
    return __builtin_bit_cast(v16bf, u);
}

// ---------------------------------------------------------------------------
// Init: convert h0 -> bf16 ping buffer, reset grid barrier (every launch).
// ---------------------------------------------------------------------------
__global__ void init_state(const float* __restrict__ state_h,
                           unsigned short* __restrict__ hbuf0,
                           unsigned int* __restrict__ bar) {
    int i = blockIdx.x * blockDim.x + threadIdx.x;
    if (i < BATCH * UNITS) hbuf0[i] = f2bf(state_h[i]);
    if (i == 0) *bar = 0u;
}

// ---------------------------------------------------------------------------
// Persistent LSTM kernel: 64 WGs x 256 threads. WG g owns units [16g, 16g+16).
// ---------------------------------------------------------------------------
__global__ __launch_bounds__(NTHREAD, 1)
void lstm_persistent(const int*   __restrict__ tokens,
                     const float* __restrict__ state_h,
                     const float* __restrict__ state_c,
                     const float* __restrict__ emb_table,
                     const float* __restrict__ W,
                     const float* __restrict__ U,
                     const float* __restrict__ bias,
                     float*       __restrict__ out,
                     unsigned short* __restrict__ hbuf0,
                     unsigned short* __restrict__ hbuf1,
                     unsigned int* __restrict__ bar) {
    extern __shared__ unsigned short smem[];
    unsigned short* Uswz = smem;                   // 4*32*32*16 u16 = 128 KB
    unsigned short* Wswz = smem + 65536;           // 4* 4*32*16 u16 =  16 KB
    unsigned short* x_s  = smem + 65536 + 8192;    // 128*128    u16 =  32 KB
    float*          xf32 = (float*)(smem + 90112); // 128*100    f32 =  50 KB
    int* tok_s = (int*)((char*)smem + 231424);     // 128 ints

    const int tid  = threadIdx.x;
    const int wg   = blockIdx.x;
    const int lane = tid & 31;
    const int w    = tid >> 5;            // wave id == M-tile id
    const int n    = lane & 15;           // column within 16-wide tile
    const int hi8  = (lane & 16) ? 8 : 0; // C-layout row offset
    const int unit = wg * 16 + n;
    const int arow = 16 * w + n;          // this wave's A-fragment row for this lane
    const unsigned xf32_base = lds_off(xf32);

    // ---- one-time: swizzle U and W slices into LDS in B-fragment layout ----
    for (int idx = tid; idx < NGATE * 32 * 32 * 16; idx += NTHREAD) {
        int e  = idx & 15;
        int ln = (idx >> 4) & 31;
        int kt = (idx >> 9) & 31;
        int g  = idx >> 14;
        int K   = kt * 32 + ((e & 8) << 1) + ((ln & 16) ? 8 : 0) + (e & 7);
        int col = g * UNITS + wg * 16 + (ln & 15);
        Uswz[idx] = f2bf(U[(size_t)K * ZDIM + col]);
    }
    for (int idx = tid; idx < NGATE * 4 * 32 * 16; idx += NTHREAD) {
        int e  = idx & 15;
        int ln = (idx >> 4) & 31;
        int kt = (idx >> 9) & 3;
        int g  = idx >> 11;
        int K   = kt * 32 + ((e & 8) << 1) + ((ln & 16) ? 8 : 0) + (e & 7);
        int col = g * UNITS + wg * 16 + (ln & 15);
        float v = (K < EMB) ? W[(size_t)K * ZDIM + col] : 0.0f;
        Wswz[idx] = f2bf(v);
    }

    // ---- persistent per-lane state: bias, h-tile, c-tile (C layout) ----
    float bv[4];
#pragma unroll
    for (int g = 0; g < 4; ++g) bv[g] = bias[g * UNITS + unit];

    float hval[8], cval[8];
#pragma unroll
    for (int r = 0; r < 8; ++r) {
        int b = 16 * w + r + hi8;
        hval[r] = state_h[(size_t)b * UNITS + unit];
        cval[r] = state_c[(size_t)b * UNITS + unit];
    }

    // ---- prime x_s / tok_s for t = 0 (synchronous) ----
    if (tid < BATCH) tok_s[tid] = tokens[(size_t)tid * TLEN + 0];
    {
        int b   = tid >> 1;
        int k0  = (tid & 1) * 64;
        int tok = tokens[(size_t)b * TLEN + 0];
        const float* er = emb_table + (size_t)tok * EMB;
#pragma unroll 8
        for (int k = k0; k < k0 + 64; ++k)
            x_s[b * 128 + k] = (k < EMB) ? f2bf(er[k]) : (unsigned short)0;
    }
    __syncthreads();

    const size_t hT_off = (size_t)BATCH * TLEN * UNITS;
    const size_t cT_off = hT_off + (size_t)BATCH * UNITS;

    for (int t = 0; t < TLEN; ++t) {
        v8f acc[4];
#pragma unroll
        for (int g = 0; g < 4; ++g) acc[g] = (v8f){0, 0, 0, 0, 0, 0, 0, 0};

        // ---- x_t @ W  (K = 128 padded) ----
#pragma unroll
        for (int kt = 0; kt < 4; ++kt) {
            v16bf a = load_a16(x_s, arow, 128, kt, lane);
#pragma unroll
            for (int g = 0; g < 4; ++g) {
                const v16bf bm = *(const v16bf*)(Wswz + (((g * 4 + kt) * 32) + lane) * 16);
                acc[g] = __builtin_amdgcn_wmma_f32_16x16x32_bf16(
                    false, a, false, bm, (short)0, acc[g], false, false);
            }
        }

        // ---- kick off async gather of x_{t+1}: 3200 aligned 16B chunks ----
#if USE_ASYNC
        if (t + 1 < TLEN) {
            for (int c = tid; c < BATCH * 25; c += NTHREAD) {
                int b  = c / 25;
                int cw = c - b * 25;
                int tok = tokens[(size_t)b * TLEN + (t + 1)];
                const float* g = emb_table + (size_t)tok * EMB + cw * 4;
                unsigned loff = xf32_base + (unsigned)(b * 400 + cw * 16);
                asm volatile("global_load_async_to_lds_b128 %0, %1, off"
                             :: "v"(loff), "v"((unsigned long long)(size_t)g)
                             : "memory");
            }
        }
#endif

        // ---- h @ U  (K = 1024): software-pipelined.
        //      A (global h, long latency): prefetch distance 2.
        //      B (LDS U-fragments):        prefetch distance 1.
        const unsigned short* hcur  = (t & 1) ? hbuf1 : hbuf0;
        unsigned short*       hnext = (t & 1) ? hbuf0 : hbuf1;
        {
            v16bf a0 = load_a16(hcur, arow, UNITS, 0, lane);
            v16bf a1 = load_a16(hcur, arow, UNITS, 1, lane);
            v16bf bc[4];
#pragma unroll
            for (int g = 0; g < 4; ++g)
                bc[g] = *(const v16bf*)(Uswz + (((g * 32) * 32) + lane) * 16);

#pragma unroll 4
            for (int kt = 0; kt < 32; ++kt) {
                const int kt1 = (kt + 1) & 31;   // wrap: tail prefetches are harmless
                const int kt2 = (kt + 2) & 31;
                v16bf a2 = load_a16(hcur, arow, UNITS, kt2, lane);
                v16bf bn[4];
#pragma unroll
                for (int g = 0; g < 4; ++g)
                    bn[g] = *(const v16bf*)(Uswz + (((g * 32 + kt1) * 32) + lane) * 16);
#pragma unroll
                for (int g = 0; g < 4; ++g)
                    acc[g] = __builtin_amdgcn_wmma_f32_16x16x32_bf16(
                        false, a0, false, bc[g], (short)0, acc[g], false, false);
                a0 = a1;
                a1 = a2;
#pragma unroll
                for (int g = 0; g < 4; ++g) bc[g] = bn[g];
            }
        }

        // ---- gates, state update, masked write-back ----
#pragma unroll
        for (int r = 0; r < 8; ++r) {
            int b = 16 * w + r + hi8;
            bool m = tok_s[b] != 0;
            float zi = acc[0][r] + bv[0];
            float zf = acc[1][r] + bv[1];
            float zg = acc[2][r] + bv[2];
            float zo = acc[3][r] + bv[3];
            float cn = sigf(zf) * cval[r] + sigf(zi) * sigf(zg);
            float hn = sigf(zo) * sigf(cn);
            if (m) { cval[r] = cn; hval[r] = hn; }
            out[(size_t)b * TLEN * UNITS + (size_t)t * UNITS + unit] = hval[r];
            hnext[(size_t)b * UNITS + unit] = f2bf(hval[r]);
        }

        // ---- land the async gather, convert f32 -> bf16 staging for t+1 ----
        if (t + 1 < TLEN) {
#if USE_ASYNC
            asm volatile("s_wait_asynccnt 0x0" ::: "memory");
            __syncthreads();   // all waves done with x_s(t) and gate math
            {
                int b  = tid >> 1;
                int k0 = (tid & 1) * 64;
                const float* xr = xf32 + (size_t)b * EMB;
#pragma unroll 8
                for (int k = k0; k < k0 + 64; ++k)
                    x_s[b * 128 + k] = (k < EMB) ? f2bf(xr[k]) : (unsigned short)0;
            }
#else
            __syncthreads();
            {
                int b   = tid >> 1;
                int k0  = (tid & 1) * 64;
                int tok = tokens[(size_t)b * TLEN + (t + 1)];
                const float* er = emb_table + (size_t)tok * EMB;
#pragma unroll 8
                for (int k = k0; k < k0 + 64; ++k)
                    x_s[b * 128 + k] = (k < EMB) ? f2bf(er[k]) : (unsigned short)0;
            }
#endif
            if (tid < BATCH) tok_s[tid] = tokens[(size_t)tid * TLEN + (t + 1)];
        }

        // ---- grid-wide step barrier (monotonic counter, reset by init) ----
        __threadfence();
        __syncthreads();
        if (tid == 0) {
            atomicAdd(bar, 1u);
            unsigned target = (unsigned)(t + 1) * NWG;
            while (__hip_atomic_load(bar, __ATOMIC_ACQUIRE, __HIP_MEMORY_SCOPE_AGENT) < target)
                __builtin_amdgcn_s_sleep(1);
        }
        __syncthreads();
    }

    // ---- final states hT, cT ----
#pragma unroll
    for (int r = 0; r < 8; ++r) {
        int b = 16 * w + r + hi8;
        out[hT_off + (size_t)b * UNITS + unit] = hval[r];
        out[cT_off + (size_t)b * UNITS + unit] = cval[r];
    }
}

// ---------------------------------------------------------------------------
extern "C" void kernel_launch(void* const* d_in, const int* in_sizes, int n_in,
                              void* d_out, int out_size, void* d_ws, size_t ws_size,
                              hipStream_t stream) {
    const int*   tokens  = (const int*)  d_in[0];
    const float* state_h = (const float*)d_in[1];
    const float* state_c = (const float*)d_in[2];
    const float* emb     = (const float*)d_in[3];
    const float* W       = (const float*)d_in[4];
    const float* U       = (const float*)d_in[5];
    const float* bias    = (const float*)d_in[6];
    float* out = (float*)d_out;

    unsigned char*  ws    = (unsigned char*)d_ws;
    unsigned int*   bar   = (unsigned int*)ws;                          // 256 B slot
    unsigned short* hbuf0 = (unsigned short*)(ws + 256);                // 256 KB
    unsigned short* hbuf1 = (unsigned short*)(ws + 256 + BATCH * UNITS * 2);

    init_state<<<(BATCH * UNITS + 255) / 256, 256, 0, stream>>>(state_h, hbuf0, bar);

    size_t lds_bytes = 231424 + 512;   // Uswz+Wswz+x_s + xf32 + tok_s ≈ 226.5 KB
    lstm_persistent<<<NWG, NTHREAD, lds_bytes, stream>>>(
        tokens, state_h, state_c, emb, W, U, bias, out, hbuf0, hbuf1, bar);
}
